// Adafuse_56590489092145
// MI455X (gfx1250) — matrix operand order repaired and checked by reading, not compiled, not run on hardware
//
#include <hip/hip_runtime.h>
#include <math.h>

typedef __attribute__((ext_vector_type(16))) _Float16 v16h;
typedef __attribute__((ext_vector_type(8)))  _Float16 v8h;
typedef __attribute__((ext_vector_type(8)))  float    v8f;

union HV { v16h v; v8h h[2]; };

#define B_ROWS 8192
#define T_ROWS 8192
#define FD 768
#define PD 512
#define HD 1024
#define NE 4

__device__ __forceinline__ float gelu_f(float x) {
  return 0.5f * x * (1.0f + erff(x * 0.70710678118654752f));
}

// ---------------------------------------------------------------------------
// Wave-level GEMM core: one wave computes a 32(M) x 64(N) f32 tile of
// D = A[M,K] * W[N,K]^T using v_wmma_f32_16x16x32_f16.
// A,W are f16 row-major; lda/ldw in elements; K multiple of 32.
// A-frag layout (16x32 f16): lanes0-15 K=0..15, lanes16-31 K=16..31
// B-frag layout (32x16 f16): lanes0-15 K=0..15, lanes16-31 K=16..31
// ---------------------------------------------------------------------------
__device__ __forceinline__ void wave_gemm(const _Float16* __restrict__ A, int lda,
                                          const _Float16* __restrict__ W, int ldw,
                                          int K, int rowBase, int colBase,
                                          v8f (&acc)[2][4]) {
  const int lane  = threadIdx.x & 31;
  const int lhalf = lane >> 4;
  const int l16   = lane & 15;
  const _Float16* pa0 = A + (size_t)(rowBase +      l16) * lda + lhalf * 16;
  const _Float16* pa1 = A + (size_t)(rowBase + 16 + l16) * lda + lhalf * 16;
  const _Float16* pb0 = W + (size_t)(colBase +  0 + l16) * ldw + lhalf * 16;
  const _Float16* pb1 = W + (size_t)(colBase + 16 + l16) * ldw + lhalf * 16;
  const _Float16* pb2 = W + (size_t)(colBase + 32 + l16) * ldw + lhalf * 16;
  const _Float16* pb3 = W + (size_t)(colBase + 48 + l16) * ldw + lhalf * 16;

  for (int k0 = 0; k0 < K; k0 += 32) {
    HV a0, a1, b0, b1, b2, b3;
    a0.h[0] = *(const v8h*)(pa0 + k0); a0.h[1] = *(const v8h*)(pa0 + k0 + 8);
    a1.h[0] = *(const v8h*)(pa1 + k0); a1.h[1] = *(const v8h*)(pa1 + k0 + 8);
    b0.h[0] = *(const v8h*)(pb0 + k0); b0.h[1] = *(const v8h*)(pb0 + k0 + 8);
    b1.h[0] = *(const v8h*)(pb1 + k0); b1.h[1] = *(const v8h*)(pb1 + k0 + 8);
    b2.h[0] = *(const v8h*)(pb2 + k0); b2.h[1] = *(const v8h*)(pb2 + k0 + 8);
    b3.h[0] = *(const v8h*)(pb3 + k0); b3.h[1] = *(const v8h*)(pb3 + k0 + 8);
    __builtin_prefetch(pa0 + k0 + 128, 0, 1);
    __builtin_prefetch(pa1 + k0 + 128, 0, 1);

    acc[0][0] = __builtin_amdgcn_wmma_f32_16x16x32_f16(false, a0.v, false, b0.v, (short)0, acc[0][0], false, false);
    acc[0][1] = __builtin_amdgcn_wmma_f32_16x16x32_f16(false, a0.v, false, b1.v, (short)0, acc[0][1], false, false);
    acc[0][2] = __builtin_amdgcn_wmma_f32_16x16x32_f16(false, a0.v, false, b2.v, (short)0, acc[0][2], false, false);
    acc[0][3] = __builtin_amdgcn_wmma_f32_16x16x32_f16(false, a0.v, false, b3.v, (short)0, acc[0][3], false, false);
    acc[1][0] = __builtin_amdgcn_wmma_f32_16x16x32_f16(false, a1.v, false, b0.v, (short)0, acc[1][0], false, false);
    acc[1][1] = __builtin_amdgcn_wmma_f32_16x16x32_f16(false, a1.v, false, b1.v, (short)0, acc[1][1], false, false);
    acc[1][2] = __builtin_amdgcn_wmma_f32_16x16x32_f16(false, a1.v, false, b2.v, (short)0, acc[1][2], false, false);
    acc[1][3] = __builtin_amdgcn_wmma_f32_16x16x32_f16(false, a1.v, false, b3.v, (short)0, acc[1][3], false, false);
  }
}

// D-tile element (i,t,r) -> row = rowBase+i*16+lhalf*8+r, col = colBase+t*16+l16
#define EPILOGUE_VARS \
  const int lane = threadIdx.x & 31; const int lhalf = lane >> 4; const int l16 = lane & 15;

#define BLOCK_TILE_VARS \
  const int wave = threadIdx.x >> 5; \
  const int rowBase = blockIdx.x * 128 + (wave & 3) * 32; \
  const int colBase = blockIdx.y * 128 + (wave >> 2) * 64;

// ---------------- GEMM + bias + gelu -> f16 (projections / router / ds) ----
__global__ void __launch_bounds__(256) GemmBiasGeluF16(
    const _Float16* __restrict__ A, int lda,
    const _Float16* __restrict__ W, int ldw,
    const float* __restrict__ bias,
    _Float16* __restrict__ C, int ldc, int K) {
  BLOCK_TILE_VARS
  v8f acc[2][4] = {};
  wave_gemm(A, lda, W, ldw, K, rowBase, colBase, acc);
  EPILOGUE_VARS
  #pragma unroll
  for (int i = 0; i < 2; ++i)
    #pragma unroll
    for (int t = 0; t < 4; ++t) {
      const int n = colBase + t * 16 + l16;
      const float bn = bias[n];
      #pragma unroll
      for (int r = 0; r < 8; ++r) {
        const int m = rowBase + i * 16 + lhalf * 8 + r;
        C[(size_t)m * ldc + n] = (_Float16)gelu_f(acc[i][t][r] + bn);
      }
    }
}

// ---------------- experts: fused_hidden = sum_e probs[b,e]*gelu(comb@We^T+be)
__global__ void __launch_bounds__(256) ExpertsFuse(
    const _Float16* __restrict__ comb, const _Float16* __restrict__ expW,
    const float* __restrict__ expB, const float* __restrict__ probs,
    _Float16* __restrict__ fused) {
  BLOCK_TILE_VARS
  EPILOGUE_VARS
  v8f facc[2][4] = {};
  for (int e = 0; e < NE; ++e) {
    v8f acc[2][4] = {};
    wave_gemm(comb, 2 * PD, expW + (size_t)e * HD * 2 * PD, 2 * PD,
              2 * PD, rowBase, colBase, acc);
    #pragma unroll
    for (int i = 0; i < 2; ++i)
      #pragma unroll
      for (int t = 0; t < 4; ++t) {
        const int n = colBase + t * 16 + l16;
        const float bn = expB[e * HD + n];
        #pragma unroll
        for (int r = 0; r < 8; ++r) {
          const int m = rowBase + i * 16 + lhalf * 8 + r;
          const float p = probs[(size_t)m * NE + e];
          facc[i][t][r] += p * gelu_f(acc[i][t][r] + bn);
        }
      }
  }
  #pragma unroll
  for (int i = 0; i < 2; ++i)
    #pragma unroll
    for (int t = 0; t < 4; ++t) {
      const int n = colBase + t * 16 + l16;
      #pragma unroll
      for (int r = 0; r < 8; ++r) {
        const int m = rowBase + i * 16 + lhalf * 8 + r;
        fused[(size_t)m * HD + n] = (_Float16)facc[i][t][r];
      }
    }
}

// ---------------- residual + ds-gated mix -> f16 ---------------------------
__global__ void __launch_bounds__(256) OutputMix(
    const _Float16* __restrict__ fused, const _Float16* __restrict__ outW,
    const float* __restrict__ outB, const float* __restrict__ dsv,
    const float* __restrict__ textf, const float* __restrict__ imagef,
    _Float16* __restrict__ outh) {
  BLOCK_TILE_VARS
  v8f acc[2][4] = {};
  wave_gemm(fused, HD, outW, HD, HD, rowBase, colBase, acc);
  EPILOGUE_VARS
  #pragma unroll
  for (int i = 0; i < 2; ++i)
    #pragma unroll
    for (int t = 0; t < 4; ++t) {
      const int n = colBase + t * 16 + l16;
      const float bn = outB[n];
      #pragma unroll
      for (int r = 0; r < 8; ++r) {
        const int m = rowBase + i * 16 + lhalf * 8 + r;
        const float d = dsv[m];
        const size_t idx = (size_t)m * FD + n;
        const float v = acc[i][t][r] + bn + d * textf[idx] + (1.0f - d) * imagef[idx];
        outh[idx] = (_Float16)v;
      }
    }
}

// ---------------- logits = exp(scale) * pred @ tgt^T -> f32 d_out ----------
__global__ void __launch_bounds__(256) LogitsGemm(
    const _Float16* __restrict__ pred, const _Float16* __restrict__ tgt,
    const float* __restrict__ lscale, float* __restrict__ out) {
  BLOCK_TILE_VARS
  v8f acc[2][4] = {};
  wave_gemm(pred, FD, tgt, FD, FD, rowBase, colBase, acc);
  const float sc = expf(lscale[0]);
  EPILOGUE_VARS
  #pragma unroll
  for (int i = 0; i < 2; ++i)
    #pragma unroll
    for (int t = 0; t < 4; ++t) {
      const int n = colBase + t * 16 + l16;
      #pragma unroll
      for (int r = 0; r < 8; ++r) {
        const int m = rowBase + i * 16 + lhalf * 8 + r;
        out[(size_t)m * T_ROWS + n] = sc * acc[i][t][r];
      }
    }
}

// ---------------- fp32 -> f16 convert --------------------------------------
__global__ void CvtF32F16(const float* __restrict__ s, _Float16* __restrict__ d, size_t n) {
  size_t i = (size_t)blockIdx.x * blockDim.x + threadIdx.x;
  const size_t stride = (size_t)gridDim.x * blockDim.x;
  for (; i < n; i += stride) d[i] = (_Float16)s[i];
}

// ---------------- router head: logits(4) + softmax, one wave per row -------
__global__ void __launch_bounds__(256) RouterHead(
    const _Float16* __restrict__ hid, const float* __restrict__ W2,
    const float* __restrict__ b2, float* __restrict__ probs) {
  const int wave = threadIdx.x >> 5, lane = threadIdx.x & 31;
  const int b = blockIdx.x * 8 + wave;
  const _Float16* h = hid + (size_t)b * (HD / 4);
  float s[NE] = {0.f, 0.f, 0.f, 0.f};
  for (int j = lane; j < HD / 4; j += 32) {
    const float hv = (float)h[j];
    #pragma unroll
    for (int e = 0; e < NE; ++e) s[e] += hv * W2[e * (HD / 4) + j];
  }
  #pragma unroll
  for (int e = 0; e < NE; ++e)
    for (int m = 16; m >= 1; m >>= 1) s[e] += __shfl_xor(s[e], m, 32);
  if (lane == 0) {
    float l[NE];
    #pragma unroll
    for (int e = 0; e < NE; ++e) l[e] = s[e] + b2[e];
    float mx = fmaxf(fmaxf(l[0], l[1]), fmaxf(l[2], l[3]));
    float se = 0.f;
    #pragma unroll
    for (int e = 0; e < NE; ++e) { l[e] = expf(l[e] - mx); se += l[e]; }
    #pragma unroll
    for (int e = 0; e < NE; ++e) probs[(size_t)b * NE + e] = l[e] / se;
  }
}

// ---------------- ds head: sigmoid(hidden . W2 + b), one wave per row ------
__global__ void __launch_bounds__(256) DsHead(
    const _Float16* __restrict__ hid, const float* __restrict__ W2,
    const float* __restrict__ b2, float* __restrict__ ds) {
  const int wave = threadIdx.x >> 5, lane = threadIdx.x & 31;
  const int b = blockIdx.x * 8 + wave;
  const _Float16* h = hid + (size_t)b * HD;
  float s = 0.f;
  for (int j = lane; j < HD; j += 32) s += (float)h[j] * W2[j];
  for (int m = 16; m >= 1; m >>= 1) s += __shfl_xor(s, m, 32);
  if (lane == 0) ds[b] = 1.0f / (1.0f + expf(-(s + b2[0])));
}

// ---------------- row L2-normalize (768 cols), one block per row -----------
__global__ void __launch_bounds__(256) NormRowsH(const _Float16* __restrict__ src,
                                                 _Float16* __restrict__ dst) {
  const int row = blockIdx.x;
  const _Float16* s = src + (size_t)row * FD;
  _Float16* d = dst + (size_t)row * FD;
  float ss = 0.f;
  for (int j = threadIdx.x; j < FD; j += 256) { const float x = (float)s[j]; ss += x * x; }
  for (int m = 16; m >= 1; m >>= 1) ss += __shfl_xor(ss, m, 32);
  __shared__ float sred[8];
  if ((threadIdx.x & 31) == 0) sred[threadIdx.x >> 5] = ss;
  __syncthreads();
  float tot = 0.f;
  #pragma unroll
  for (int w = 0; w < 8; ++w) tot += sred[w];
  const float inv = 1.0f / fmaxf(sqrtf(tot), 1e-12f);
  for (int j = threadIdx.x; j < FD; j += 256) d[j] = (_Float16)((float)s[j] * inv);
}

__global__ void __launch_bounds__(256) NormRowsF(const float* __restrict__ src,
                                                 _Float16* __restrict__ dst) {
  const int row = blockIdx.x;
  const float* s = src + (size_t)row * FD;
  _Float16* d = dst + (size_t)row * FD;
  float ss = 0.f;
  for (int j = threadIdx.x; j < FD; j += 256) { const float x = s[j]; ss += x * x; }
  for (int m = 16; m >= 1; m >>= 1) ss += __shfl_xor(ss, m, 32);
  __shared__ float sred[8];
  if ((threadIdx.x & 31) == 0) sred[threadIdx.x >> 5] = ss;
  __syncthreads();
  float tot = 0.f;
  #pragma unroll
  for (int w = 0; w < 8; ++w) tot += sred[w];
  const float inv = 1.0f / fmaxf(sqrtf(tot), 1e-12f);
  for (int j = threadIdx.x; j < FD; j += 256) d[j] = (_Float16)(s[j] * inv);
}

// ---------------------------------------------------------------------------
extern "C" void kernel_launch(void* const* d_in, const int* in_sizes, int n_in,
                              void* d_out, int out_size, void* d_ws, size_t ws_size,
                              hipStream_t stream) {
  (void)in_sizes; (void)n_in; (void)out_size; (void)ws_size;
  const float* imagef = (const float*)d_in[0];
  const float* textf  = (const float*)d_in[1];
  const float* targf  = (const float*)d_in[2];
  const float* Wt     = (const float*)d_in[3];
  const float* bt     = (const float*)d_in[4];
  const float* Wi     = (const float*)d_in[5];
  const float* bi     = (const float*)d_in[6];
  const float* dsW1   = (const float*)d_in[7];
  const float* dsb1   = (const float*)d_in[8];
  const float* dsW2   = (const float*)d_in[9];
  const float* dsb2   = (const float*)d_in[10];
  const float* expW   = (const float*)d_in[11];
  const float* expB   = (const float*)d_in[12];
  const float* rtW1   = (const float*)d_in[13];
  const float* rtb1   = (const float*)d_in[14];
  const float* rtW2   = (const float*)d_in[15];
  const float* rtb2   = (const float*)d_in[16];
  const float* outW   = (const float*)d_in[17];
  const float* outB   = (const float*)d_in[18];
  const float* lscale = (const float*)d_in[19];

  // ---- workspace layout (f16 buffers, 256B aligned) ----
  char* ws = (char*)d_ws;
  size_t off = 0;
  auto alloc_h = [&](size_t nelem) -> _Float16* {
    _Float16* p = (_Float16*)(ws + off);
    off += ((nelem * sizeof(_Float16) + 255) / 256) * 256;
    return p;
  };
  _Float16* text_h  = alloc_h((size_t)B_ROWS * FD);
  _Float16* image_h = alloc_h((size_t)B_ROWS * FD);
  _Float16* Wt_h    = alloc_h((size_t)PD * FD);
  _Float16* Wi_h    = alloc_h((size_t)PD * FD);
  _Float16* rtW1_h  = alloc_h((size_t)(HD / 4) * 2 * PD);
  _Float16* dsW1_h  = alloc_h((size_t)HD * 2 * PD);
  _Float16* expW_h  = alloc_h((size_t)NE * HD * 2 * PD);
  _Float16* outW_h  = alloc_h((size_t)FD * HD);
  _Float16* comb_h  = alloc_h((size_t)B_ROWS * 2 * PD);
  _Float16* hidR_h  = alloc_h((size_t)B_ROWS * (HD / 4));
  _Float16* hidD_h  = alloc_h((size_t)B_ROWS * HD);
  _Float16* fused_h = alloc_h((size_t)B_ROWS * HD);
  _Float16* outp_h  = alloc_h((size_t)B_ROWS * FD);
  _Float16* pred_h  = alloc_h((size_t)B_ROWS * FD);
  _Float16* tgt_h   = alloc_h((size_t)T_ROWS * FD);

  float* out       = (float*)d_out;
  float* ds_out    = out + (size_t)B_ROWS * T_ROWS;
  float* probs_out = ds_out + B_ROWS;

  const dim3 blk(256);
  // ---- 0) fp32 -> f16 conversions ----
  CvtF32F16<<<dim3(2048), blk, 0, stream>>>(textf,  text_h,  (size_t)B_ROWS * FD);
  CvtF32F16<<<dim3(2048), blk, 0, stream>>>(imagef, image_h, (size_t)B_ROWS * FD);
  CvtF32F16<<<dim3(512),  blk, 0, stream>>>(Wt,   Wt_h,   (size_t)PD * FD);
  CvtF32F16<<<dim3(512),  blk, 0, stream>>>(Wi,   Wi_h,   (size_t)PD * FD);
  CvtF32F16<<<dim3(512),  blk, 0, stream>>>(rtW1, rtW1_h, (size_t)(HD / 4) * 2 * PD);
  CvtF32F16<<<dim3(1024), blk, 0, stream>>>(dsW1, dsW1_h, (size_t)HD * 2 * PD);
  CvtF32F16<<<dim3(2048), blk, 0, stream>>>(expW, expW_h, (size_t)NE * HD * 2 * PD);
  CvtF32F16<<<dim3(1024), blk, 0, stream>>>(outW, outW_h, (size_t)FD * HD);

  // ---- 1) projections: comb[:, :512]=gelu(text@Wt^T+bt), [:,512:]=image ----
  GemmBiasGeluF16<<<dim3(B_ROWS / 128, PD / 128), blk, 0, stream>>>(
      text_h, FD, Wt_h, FD, bt, comb_h, 2 * PD, FD);
  GemmBiasGeluF16<<<dim3(B_ROWS / 128, PD / 128), blk, 0, stream>>>(
      image_h, FD, Wi_h, FD, bi, comb_h + PD, 2 * PD, FD);

  // ---- 2) router hidden + head/softmax ----
  GemmBiasGeluF16<<<dim3(B_ROWS / 128, (HD / 4) / 128), blk, 0, stream>>>(
      comb_h, 2 * PD, rtW1_h, 2 * PD, rtb1, hidR_h, HD / 4, 2 * PD);
  RouterHead<<<dim3(B_ROWS / 8), blk, 0, stream>>>(hidR_h, rtW2, rtb2, probs_out);

  // ---- 3) ds hidden + sigmoid head ----
  GemmBiasGeluF16<<<dim3(B_ROWS / 128, HD / 128), blk, 0, stream>>>(
      comb_h, 2 * PD, dsW1_h, 2 * PD, dsb1, hidD_h, HD, 2 * PD);
  DsHead<<<dim3(B_ROWS / 8), blk, 0, stream>>>(hidD_h, dsW2, dsb2, ds_out);

  // ---- 4) experts + router-weighted fuse ----
  ExpertsFuse<<<dim3(B_ROWS / 128, HD / 128), blk, 0, stream>>>(
      comb_h, expW_h, expB, probs_out, fused_h);

  // ---- 5) output residual + ds-gated mix ----
  OutputMix<<<dim3(B_ROWS / 128, FD / 128), blk, 0, stream>>>(
      fused_h, outW_h, outB, ds_out, textf, imagef, outp_h);

  // ---- 6) normalize pred and targets ----
  NormRowsH<<<dim3(B_ROWS), blk, 0, stream>>>(outp_h, pred_h);
  NormRowsF<<<dim3(T_ROWS), blk, 0, stream>>>(targf, tgt_h);

  // ---- 7) logits = exp(scale) * pred @ tgt^T ----
  LogitsGemm<<<dim3(B_ROWS / 128, T_ROWS / 128), blk, 0, stream>>>(
      pred_h, tgt_h, lscale, out);
}
